// ESNPyTorch_14877766713618
// MI455X (gfx1250) — matrix operand order repaired and checked by reading, not compiled, not run on hardware
//
#include <hip/hip_runtime.h>
#include <hip/hip_fp16.h>
#include <math.h>
#include <stdint.h>

// ---------------------------------------------------------------------------
// ESN leaky scan for MI455X (gfx1250, wave32).
//   x[t+1] = 0.9*x[t] + 0.1*tanh(W @ x[t] + Win @ u[t] + b)
// Roofline: W@x matvec dominates (512 * 128MB f16 reads, L2-resident since
// 128MB < 192MB L2).  Win@u precompute is a real GEMM -> v_wmma_f32_16x16x32_f16.
// x staging uses the CDNA5 async Global->LDS path (ASYNCcnt).
// ---------------------------------------------------------------------------

typedef _Float16 h16v __attribute__((ext_vector_type(16)));
typedef _Float16 h8v  __attribute__((ext_vector_type(8)));
typedef _Float16 h4v  __attribute__((ext_vector_type(4)));
typedef float    f8v  __attribute__((ext_vector_type(8)));
typedef float    f4v  __attribute__((ext_vector_type(4)));

#define N_STATE 8192
#define D_IN    256
#define T_STEPS 512
#define LR      0.1f

// ---- fp32 -> f16 converter, 4 elements per thread (global_load_b128) -------
__global__ void cvt4_f32_to_f16(const float* __restrict__ in,
                                _Float16* __restrict__ out, int n4) {
    int i = blockIdx.x * blockDim.x + threadIdx.x;
    int stride = gridDim.x * blockDim.x;
    for (; i < n4; i += stride) {
        f4v v = ((const f4v*)in)[i];
        h4v h;
        h[0] = (_Float16)v[0]; h[1] = (_Float16)v[1];
        h[2] = (_Float16)v[2]; h[3] = (_Float16)v[3];
        ((h4v*)out)[i] = h;
    }
}

__global__ void zero_f32(float* __restrict__ p, int n) {
    int i = blockIdx.x * blockDim.x + threadIdx.x;
    if (i < n) p[i] = 0.f;
}

// ---- U[t][n] = b[n] + sum_k Win[n][k]*u[t][k]  via WMMA f16 ----------------
// One wave computes a 16(n) x 16(t) tile of U; K=256 in 8 steps of 32.
// A-frag (16x32 f16): lane&15 = row n, elems 0..15 = K (lane>>4)*16 .. +15.
// B-frag (32x16 f16): lane&15 = column t, elems 0..15 = K (lane>>4)*16 .. +15.
__global__ __launch_bounds__(256)
void uproj_wmma(const _Float16* __restrict__ Winh,  // [N_STATE, D_IN]
                const _Float16* __restrict__ uh,    // [T_STEPS, D_IN]
                const float*    __restrict__ b,     // [N_STATE]
                float*          __restrict__ U)     // [T_STEPS, N_STATE]
{
    int waveId = (blockIdx.x * blockDim.x + threadIdx.x) >> 5;
    int lane   = threadIdx.x & 31;
    int tileN  = waveId & (N_STATE / 16 - 1);   // 512 tiles along n
    int tileT  = waveId >> 9;                   // 32 tiles along t
    int n0 = tileN * 16, t0 = tileT * 16;
    int l15 = lane & 15;
    int hi  = lane >> 4;

    const _Float16* arow = Winh + (size_t)(n0 + l15) * D_IN + hi * 16;
    const _Float16* brow = uh   + (size_t)(t0 + l15) * D_IN + hi * 16;

    f8v acc = {};
    #pragma unroll
    for (int k0 = 0; k0 < D_IN; k0 += 32) {
        h8v alo = *(const h8v*)(arow + k0);        // K = hi*16 + 0..7
        h8v ahi = *(const h8v*)(arow + k0 + 8);    // K = hi*16 + 8..15
        h16v a;
        #pragma unroll
        for (int i = 0; i < 8; ++i) { a[i] = alo[i]; a[8 + i] = ahi[i]; }
        h16v bf = *(const h16v*)(brow + k0);       // K = hi*16 + 0..15
        acc = __builtin_amdgcn_wmma_f32_16x16x32_f16(
            /*neg_a=*/false, a, /*neg_b=*/false, bf,
            /*c_mod=*/(short)0, acc, /*reuse_a=*/false, /*reuse_b=*/false);
    }

    // D layout: elem j -> row M = j + hi*8 (n), col = l15 (t)
    int t = t0 + l15;
    #pragma unroll
    for (int j = 0; j < 8; ++j) {
        int n = n0 + hi * 8 + j;
        U[(size_t)t * N_STATE + n] = acc[j] + b[n];
    }
}

// ---- one recurrence step: y = W@x ; x' = 0.9 x + 0.1 tanh(y + U[t]) --------
// 256 blocks x 256 threads; 32 rows/block, 8 lanes/row (wave32 -> 4 rows/wave).
// x staged in LDS via global_load_async_to_lds_b128 (ASYNCcnt path); lane s
// reads x[base + s*8 .. +7] -> 8 lanes span exactly 256 B = all 64 banks,
// conflict-free; lanes with equal s broadcast.
__global__ __launch_bounds__(256)
void esn_step(const _Float16* __restrict__ Wh,    // [N_STATE, N_STATE] f16
              const float*    __restrict__ Urow,  // U + t*N
              const float*    __restrict__ xprev, // x[t]   (N floats)
              float*          __restrict__ xnext) // x[t+1] == states[t]
{
    __shared__ float xs[N_STATE];
    const int tid = threadIdx.x;

    // Async-stage x into LDS: each lane copies 16 B per round, 8 rounds.
    // LDS operand of global_load_async_to_lds is the 32-bit LDS byte offset;
    // for a generic LDS pointer that is its low 32 bits (aperture scheme).
    #pragma unroll
    for (int i = 0; i < N_STATE / 4 / 256; ++i) {
        uint32_t lds = (uint32_t)(uintptr_t)(xs + (size_t)(tid + i * 256) * 4);
        uint64_t ga  = (uint64_t)(uintptr_t)(xprev + (size_t)(tid + i * 256) * 4);
        asm volatile("global_load_async_to_lds_b128 %0, %1, off"
                     :: "v"(lds), "v"(ga) : "memory");
    }
    asm volatile("s_wait_asynccnt 0x0" ::: "memory");
    __syncthreads();

    const int lane = tid & 31;
    const int wave = tid >> 5;
    const int s    = lane & 7;
    const int row  = blockIdx.x * 32 + wave * 4 + (lane >> 3);

    const _Float16* wrow = Wh + (size_t)row * N_STATE + s * 8;
    float acc = 0.f;
    #pragma unroll 4
    for (int base = 0; base < N_STATE; base += 64) {
        __builtin_prefetch(wrow + base + 4096, 0, 0);   // global_prefetch_b8
        h8v w = *(const h8v*)(wrow + base);             // 8 f16 weights, 16B
        const float* xp = xs + base + s * 8;
        acc += (float)w[0] * xp[0] + (float)w[1] * xp[1]
             + (float)w[2] * xp[2] + (float)w[3] * xp[3]
             + (float)w[4] * xp[4] + (float)w[5] * xp[5]
             + (float)w[6] * xp[6] + (float)w[7] * xp[7];
    }

    // reduce the 8 partial sums of each row (lanes s = 0..7)
    acc += __shfl_xor(acc, 4, 32);
    acc += __shfl_xor(acc, 2, 32);
    acc += __shfl_xor(acc, 1, 32);

    if (s == 0) {
        float pre = acc + Urow[row];
        xnext[row] = (1.0f - LR) * xprev[row] + LR * tanhf(pre);
    }
}

// ---------------------------------------------------------------------------
extern "C" void kernel_launch(void* const* d_in, const int* in_sizes, int n_in,
                              void* d_out, int out_size, void* d_ws, size_t ws_size,
                              hipStream_t stream) {
    (void)in_sizes; (void)n_in; (void)out_size; (void)ws_size;
    const float* u   = (const float*)d_in[0];   // [T, D_in]
    const float* W   = (const float*)d_in[1];   // [N, N]
    const float* Win = (const float*)d_in[2];   // [N, D_in]
    const float* b   = (const float*)d_in[3];   // [N]
    float* states = (float*)d_out;              // [T, N]

    // workspace layout (needs ~148.3 MB)
    char* ws = (char*)d_ws;
    _Float16* Wh   = (_Float16*)(ws);                            // 128 MB
    _Float16* Winh = (_Float16*)(ws + (size_t)134217728);        // 4 MB
    _Float16* uh   = (_Float16*)(ws + (size_t)138412032);        // 256 KB
    float*    U    = (float*)   (ws + (size_t)138674176);        // 16 MB
    float*    x0   = (float*)   (ws + (size_t)155451392);        // 32 KB

    // 1) convert to f16 (W becomes L2-resident: 128 MB < 192 MB)
    {
        int n4 = N_STATE * N_STATE / 4;
        cvt4_f32_to_f16<<<(n4 + 255) / 256, 256, 0, stream>>>(W, Wh, n4);
    }
    {
        int n4 = N_STATE * D_IN / 4;
        cvt4_f32_to_f16<<<(n4 + 255) / 256, 256, 0, stream>>>(Win, Winh, n4);
    }
    {
        int n4 = T_STEPS * D_IN / 4;
        cvt4_f32_to_f16<<<(n4 + 255) / 256, 256, 0, stream>>>(u, uh, n4);
    }
    zero_f32<<<N_STATE / 256, 256, 0, stream>>>(x0, N_STATE);

    // 2) U = Win @ u^T + b via WMMA: 16384 wave-tiles, 8 waves/block
    uproj_wmma<<<(N_STATE / 16) * (T_STEPS / 16) / 8, 256, 0, stream>>>(
        Winh, uh, b, U);

    // 3) sequential scan: step t reads states[t-1] (or x0), writes states[t].
    //    Stream ordering enforces the recurrence dependency.
    for (int t = 0; t < T_STEPS; ++t) {
        const float* xp = (t == 0) ? x0 : states + (size_t)(t - 1) * N_STATE;
        esn_step<<<N_STATE / 32, 256, 0, stream>>>(
            Wh, U + (size_t)t * N_STATE, xp, states + (size_t)t * N_STATE);
    }
}
